// MSARowAttentionWithPairBias_5368709120300
// MI455X (gfx1250) — matrix operand order repaired and checked by reading, not compiled, not run on hardware
//
#include <hip/hip_runtime.h>
#include <hip/hip_bf16.h>

// ---------------------------------------------------------------------------
// MSARowAttentionWithPairBias on gfx1250 (MI455X).
// All matmuls run on v_wmma_f32_16x16x32_bf16 (wave32 WMMA). bf16 inputs with
// f32 accumulation; flash-style online softmax keeps logits on-chip.
// ---------------------------------------------------------------------------

typedef __attribute__((ext_vector_type(16))) __bf16 v16bf;
typedef __attribute__((ext_vector_type(8)))  float  v8f;

union FragU { unsigned u[8]; v16bf v; };

static __device__ __forceinline__ unsigned short f2bf(float f) {
    unsigned u = __float_as_uint(f);
    unsigned r = (u + 0x7FFFu + ((u >> 16) & 1u)) >> 16;   // round-nearest-even
    return (unsigned short)r;
}
static __device__ __forceinline__ float bf2f(unsigned short b) {
    return __uint_as_float(((unsigned)b) << 16);
}

// A fragment (16x32 bf16, row-major source, leading dim = ld elements).
// ISA layout: lane&15 = M row; VGPR v holds K pair at
//   k = (v>=4)*16 + (lane>=16)*8 + (v&3)*2
static __device__ __forceinline__ v16bf load_frag_a(const unsigned short* base,
                                                    int ld, int lane) {
    FragU f;
    int row = lane & 15;
    int hi  = (lane >> 4) & 1;
#pragma unroll
    for (int v = 0; v < 8; ++v) {
        int k = ((v >= 4) ? 16 : 0) + hi * 8 + (v & 3) * 2;
        f.u[v] = *(const unsigned*)(base + row * ld + k);
    }
    return f.v;
}

// B fragment (32x16 bf16). Source is "n-major": element (k,n) at base+n*ld+k.
// ISA layout: lane&15 = N col; VGPR v holds K pair at
//   k = (v>=4)*16 + (lane>=16)*8 + (v&3)*2
static __device__ __forceinline__ v16bf load_frag_b(const unsigned short* base,
                                                    int ld, int lane) {
    FragU f;
    int col = lane & 15;
    int hi  = (lane >> 4) & 1;
#pragma unroll
    for (int v = 0; v < 8; ++v) {
        int k = ((v >= 4) ? 16 : 0) + hi * 8 + (v & 3) * 2;
        f.u[v] = *(const unsigned*)(base + col * ld + k);
    }
    return f.v;
}

// ---------------------------------------------------------------------------
// K0: convert 5 [256 x 256] fp32 weight matrices to bf16, transposed so the
// GEMM B-fragment reads consecutive-K pairs as single dword loads.
// arr 0=q (fold KEY_DIM^-0.5), 1=k, 2=v, 3=gate, 4=o.
// ---------------------------------------------------------------------------
__global__ __launch_bounds__(256) void wconv_kernel(
    const float* __restrict__ qw, const float* __restrict__ kw,
    const float* __restrict__ vw, const float* __restrict__ gw,
    const float* __restrict__ ow,
    unsigned short* __restrict__ qwT, unsigned short* __restrict__ kwT,
    unsigned short* __restrict__ vwT, unsigned short* __restrict__ gwT,
    unsigned short* __restrict__ owT) {
    int tid = blockIdx.x * 256 + threadIdx.x;     // 5 * 65536 threads
    int arr = tid >> 16;
    int r   = tid & 65535;
    int k   = r >> 8;
    int n   = r & 255;
    const float* src; unsigned short* dst; float sc = 1.0f;
    switch (arr) {
        case 0: src = qw; dst = qwT; sc = 0.17677669529663687f; break; // 1/sqrt(32)
        case 1: src = kw; dst = kwT; break;
        case 2: src = vw; dst = vwT; break;
        case 3: src = gw; dst = gwT; break;
        default: src = ow; dst = owT; break;
    }
    dst[n * 256 + k] = f2bf(src[k * 256 + n] * sc);
}

// ---------------------------------------------------------------------------
// K1: LayerNorm over 256 channels, one wave per row, bf16 output.
// ---------------------------------------------------------------------------
__global__ __launch_bounds__(256) void ln_msa_kernel(
    const float* __restrict__ x, const float* __restrict__ sc,
    const float* __restrict__ of, unsigned short* __restrict__ out) {
    int lane = threadIdx.x & 31;
    int warp = threadIdx.x >> 5;
    int row  = blockIdx.x * 8 + warp;             // 65536 rows
    const float4* xr = (const float4*)(x + (long)row * 256);
    float4 a = xr[lane * 2 + 0];
    float4 b = xr[lane * 2 + 1];
    float sum = a.x + a.y + a.z + a.w + b.x + b.y + b.z + b.w;
    float sq  = a.x*a.x + a.y*a.y + a.z*a.z + a.w*a.w
              + b.x*b.x + b.y*b.y + b.z*b.z + b.w*b.w;
#pragma unroll
    for (int m = 1; m < 32; m <<= 1) {
        sum += __shfl_xor(sum, m, 32);
        sq  += __shfl_xor(sq,  m, 32);
    }
    float mu  = sum * (1.0f / 256.0f);
    float var = sq * (1.0f / 256.0f) - mu * mu;
    float rin = rsqrtf(var + 1e-5f);
    int c0 = lane * 8;
    float v0 = (a.x - mu) * rin * sc[c0+0] + of[c0+0];
    float v1 = (a.y - mu) * rin * sc[c0+1] + of[c0+1];
    float v2 = (a.z - mu) * rin * sc[c0+2] + of[c0+2];
    float v3 = (a.w - mu) * rin * sc[c0+3] + of[c0+3];
    float v4 = (b.x - mu) * rin * sc[c0+4] + of[c0+4];
    float v5 = (b.y - mu) * rin * sc[c0+5] + of[c0+5];
    float v6 = (b.z - mu) * rin * sc[c0+6] + of[c0+6];
    float v7 = (b.w - mu) * rin * sc[c0+7] + of[c0+7];
    uint4 st;
    st.x = (unsigned)f2bf(v0) | ((unsigned)f2bf(v1) << 16);
    st.y = (unsigned)f2bf(v2) | ((unsigned)f2bf(v3) << 16);
    st.z = (unsigned)f2bf(v4) | ((unsigned)f2bf(v5) << 16);
    st.w = (unsigned)f2bf(v6) | ((unsigned)f2bf(v7) << 16);
    ((uint4*)(out + (long)row * 256))[lane] = st;
}

// ---------------------------------------------------------------------------
// K2: pair LayerNorm (128 ch) + 8-head bias projection. One wave per (q,k).
// nb layout [h][q*256+k] fp32 (2 MB, L2-resident for the attention pass).
// ---------------------------------------------------------------------------
__global__ __launch_bounds__(256) void ln_pair_kernel(
    const float* __restrict__ p, const float* __restrict__ sc,
    const float* __restrict__ of, const float* __restrict__ w2d,
    float* __restrict__ nb) {
    int lane = threadIdx.x & 31;
    int warp = threadIdx.x >> 5;
    int pid  = blockIdx.x * 8 + warp;             // 65536 pairs
    const float4* xr = (const float4*)(p + (long)pid * 128);
    float4 a = xr[lane];
    float sum = a.x + a.y + a.z + a.w;
    float sq  = a.x*a.x + a.y*a.y + a.z*a.z + a.w*a.w;
#pragma unroll
    for (int m = 1; m < 32; m <<= 1) {
        sum += __shfl_xor(sum, m, 32);
        sq  += __shfl_xor(sq,  m, 32);
    }
    float mu  = sum * (1.0f / 128.0f);
    float var = sq * (1.0f / 128.0f) - mu * mu;
    float rin = rsqrtf(var + 1e-5f);
    int c0 = lane * 4;
    float n0 = (a.x - mu) * rin * sc[c0+0] + of[c0+0];
    float n1 = (a.y - mu) * rin * sc[c0+1] + of[c0+1];
    float n2 = (a.z - mu) * rin * sc[c0+2] + of[c0+2];
    float n3 = (a.w - mu) * rin * sc[c0+3] + of[c0+3];
    float acc[8];
#pragma unroll
    for (int h = 0; h < 8; ++h) {
        acc[h] = n0 * w2d[(c0+0)*8 + h] + n1 * w2d[(c0+1)*8 + h]
               + n2 * w2d[(c0+2)*8 + h] + n3 * w2d[(c0+3)*8 + h];
#pragma unroll
        for (int m = 1; m < 32; m <<= 1) acc[h] += __shfl_xor(acc[h], m, 32);
    }
    if (lane == 0) {
#pragma unroll
        for (int h = 0; h < 8; ++h) nb[h * 65536 + pid] = acc[h];
    }
}

// ---------------------------------------------------------------------------
// K3: bf16 WMMA GEMM  C[65536 x 256] = A[65536 x 256] * B[256 x 256].
// One wave computes a 16x16 tile in 8 k-steps of v_wmma_f32_16x16x32_bf16.
// mode 0/1: plain bf16 store [m][n]
// mode 2:   V transposed store vt[(s*256+n)*256 + r]  (s=m>>8, r=m&255)
// mode 3:   sigmoid(x + gating_b[n]) bf16 store
// ---------------------------------------------------------------------------
__global__ __launch_bounds__(256) void proj_gemm_kernel(
    const unsigned short* __restrict__ A, const unsigned short* __restrict__ BT,
    const float* __restrict__ bias, unsigned short* __restrict__ out, int mode) {
    int lane = threadIdx.x & 31;
    int warp = threadIdx.x >> 5;
    int mt = blockIdx.x * 8 + warp;               // 4096 m-tiles
    int nt = blockIdx.y;                          // 16 n-tiles
    const unsigned short* arow = A  + mt * 16 * 256;
    const unsigned short* brow = BT + nt * 16 * 256;
    v8f c = {};
#pragma unroll
    for (int ks = 0; ks < 8; ++ks) {
        v16bf a = load_frag_a(arow + ks * 32, 256, lane);
        v16bf b = load_frag_b(brow + ks * 32, 256, lane);
        c = __builtin_amdgcn_wmma_f32_16x16x32_bf16(false, a, false, b,
                                                    (short)0, c, false, false);
    }
    int col = nt * 16 + (lane & 15);
    int hi  = (lane >> 4) & 1;
#pragma unroll
    for (int i = 0; i < 8; ++i) {
        int m = mt * 16 + i + hi * 8;
        float val = c[i];
        if (mode == 3) val = 1.0f / (1.0f + __expf(-(val + bias[col])));
        if (mode == 2)
            out[(m >> 8) * 65536 + col * 256 + (m & 255)] = f2bf(val);
        else
            out[m * 256 + col] = f2bf(val);
    }
}

// ---------------------------------------------------------------------------
// K5: output GEMM, fp32 store + o_bias.
// ---------------------------------------------------------------------------
__global__ __launch_bounds__(256) void out_gemm_kernel(
    const unsigned short* __restrict__ A, const unsigned short* __restrict__ BT,
    const float* __restrict__ bias, float* __restrict__ out) {
    int lane = threadIdx.x & 31;
    int warp = threadIdx.x >> 5;
    int mt = blockIdx.x * 8 + warp;
    int nt = blockIdx.y;
    const unsigned short* arow = A  + mt * 16 * 256;
    const unsigned short* brow = BT + nt * 16 * 256;
    v8f c = {};
#pragma unroll
    for (int ks = 0; ks < 8; ++ks) {
        v16bf a = load_frag_a(arow + ks * 32, 256, lane);
        v16bf b = load_frag_b(brow + ks * 32, 256, lane);
        c = __builtin_amdgcn_wmma_f32_16x16x32_bf16(false, a, false, b,
                                                    (short)0, c, false, false);
    }
    int col = nt * 16 + (lane & 15);
    int hi  = (lane >> 4) & 1;
#pragma unroll
    for (int i = 0; i < 8; ++i) {
        int m = mt * 16 + i + hi * 8;
        out[m * 256 + col] = c[i] + bias[col];
    }
}

// ---------------------------------------------------------------------------
// K4: flash attention per (s, h, q-tile of 16 rows). KEY_DIM=32 -> one WMMA
// per 16x16 logit tile. Online softmax; P repacked C->A layout via 1KB/wave
// LDS bounce; two P*V WMMAs per 32-key step. Gate + store bf16 wavg.
// ---------------------------------------------------------------------------
__global__ __launch_bounds__(256) void attn_kernel(
    const unsigned short* __restrict__ qb, const unsigned short* __restrict__ kb,
    const unsigned short* __restrict__ vt, const unsigned short* __restrict__ gb,
    const float* __restrict__ nb, const int* __restrict__ mask,
    unsigned short* __restrict__ wavg) {
    __shared__ unsigned pbuf[8][256];             // per-wave 16x32 bf16 P tile
    int lane = threadIdx.x & 31;
    int warp = threadIdx.x >> 5;
    int wid  = blockIdx.x * 8 + warp;             // 32768 waves
    int s  = wid >> 7;
    int h  = (wid >> 4) & 7;
    int q0 = (wid & 15) * 16;
    int hi = (lane >> 4) & 1;
    int ln = lane & 15;

    // Q A-fragment (held in registers for the whole key loop)
    FragU qa;
    {
        const unsigned short* base = qb + ((s * 256 + q0) * 8 + h) * 32;
#pragma unroll
        for (int v = 0; v < 8; ++v) {
            int k = ((v >= 4) ? 16 : 0) + hi * 8 + (v & 3) * 2;
            qa.u[v] = *(const unsigned*)(base + ln * 256 + k);
        }
    }

    float mrun[8], lrun[8];
#pragma unroll
    for (int i = 0; i < 8; ++i) { mrun[i] = -1e30f; lrun[i] = 0.0f; }
    v8f oa0 = {}, oa1 = {};
    const float* nbh = nb + h * 65536;

    for (int kbk = 0; kbk < 256; kbk += 32) {
        // ---- logits: two 16x16 tiles (keys kbk..+15, kbk+16..+31) ----
        v8f s0 = {}, s1 = {};
#pragma unroll
        for (int t = 0; t < 2; ++t) {
            FragU bf;
            int key = kbk + t * 16 + ln;
            const unsigned short* base = kb + ((s * 256 + key) * 8 + h) * 32;
#pragma unroll
            for (int v = 0; v < 8; ++v)
                bf.u[v] = *(const unsigned*)(base + ((v >= 4) ? 16 : 0) + hi * 8 + (v & 3) * 2);
            if (t == 0)
                s0 = __builtin_amdgcn_wmma_f32_16x16x32_bf16(false, qa.v, false,
                         bf.v, (short)0, s0, false, false);
            else
                s1 = __builtin_amdgcn_wmma_f32_16x16x32_bf16(false, qa.v, false,
                         bf.v, (short)0, s1, false, false);
        }
        // ---- mask bias + pair bias ----
        int key0 = kbk + ln, key1 = kbk + 16 + ln;
        float mb0 = 1e9f * ((float)mask[s * 256 + key0] - 1.0f);
        float mb1 = 1e9f * ((float)mask[s * 256 + key1] - 1.0f);
#pragma unroll
        for (int i = 0; i < 8; ++i) {
            int qr = q0 + i + hi * 8;
            s0[i] += mb0 + nbh[qr * 256 + key0];
            s1[i] += mb1 + nbh[qr * 256 + key1];
        }
        // ---- online softmax (rows live across 16-lane groups) ----
        float p0[8], p1[8];
#pragma unroll
        for (int i = 0; i < 8; ++i) {
            float tm = fmaxf(s0[i], s1[i]);
#pragma unroll
            for (int m = 1; m < 16; m <<= 1) tm = fmaxf(tm, __shfl_xor(tm, m, 32));
            float mnew  = fmaxf(mrun[i], tm);
            float scale = __expf(mrun[i] - mnew);
            float e0 = __expf(s0[i] - mnew);
            float e1 = __expf(s1[i] - mnew);
            float rs = e0 + e1;
#pragma unroll
            for (int m = 1; m < 16; m <<= 1) rs += __shfl_xor(rs, m, 32);
            lrun[i] = lrun[i] * scale + rs;
            mrun[i] = mnew;
            oa0[i] *= scale;
            oa1[i] *= scale;
            p0[i] = e0; p1[i] = e1;
        }
        // ---- repack P (C layout -> A layout) through LDS ----
        unsigned short* pb = (unsigned short*)&pbuf[warp][0];
#pragma unroll
        for (int i = 0; i < 8; ++i) {
            int r = i + hi * 8;
            pb[r * 32 + ln]      = f2bf(p0[i]);
            pb[r * 32 + 16 + ln] = f2bf(p1[i]);
        }
        FragU pa;
#pragma unroll
        for (int v = 0; v < 8; ++v) {
            int k = ((v >= 4) ? 16 : 0) + hi * 8 + (v & 3) * 2;
            pa.u[v] = pbuf[warp][(ln * 32 + k) >> 1];
        }
        // ---- O += P * V  (V stored [s][h*32+f][r], key pairs contiguous) ----
        const unsigned short* vbase = vt + (s * 256 + h * 32) * 256 + kbk;
#pragma unroll
        for (int t = 0; t < 2; ++t) {
            FragU vf;
            int f = t * 16 + ln;
#pragma unroll
            for (int v = 0; v < 8; ++v)
                vf.u[v] = *(const unsigned*)(vbase + f * 256 + ((v >= 4) ? 16 : 0) + hi * 8 + (v & 3) * 2);
            if (t == 0)
                oa0 = __builtin_amdgcn_wmma_f32_16x16x32_bf16(false, pa.v, false,
                          vf.v, (short)0, oa0, false, false);
            else
                oa1 = __builtin_amdgcn_wmma_f32_16x16x32_bf16(false, pa.v, false,
                          vf.v, (short)0, oa1, false, false);
        }
    }
    // ---- normalize, gate, store bf16 weighted_avg [s][q][h*32+f] ----
#pragma unroll
    for (int i = 0; i < 8; ++i) {
        int qr  = q0 + i + hi * 8;
        float inv = 1.0f / lrun[i];
        const unsigned short* g = gb + ((s * 256 + qr) * 8 + h) * 32;
        unsigned short* o = wavg + (s * 256 + qr) * 256 + h * 32;
        o[ln]      = f2bf(oa0[i] * inv * bf2f(g[ln]));
        o[16 + ln] = f2bf(oa1[i] * inv * bf2f(g[16 + ln]));
    }
}

// ---------------------------------------------------------------------------
extern "C" void kernel_launch(void* const* d_in, const int* in_sizes, int n_in,
                              void* d_out, int out_size, void* d_ws, size_t ws_size,
                              hipStream_t stream) {
    const float* msa_act  = (const float*)d_in[0];
    const int*   msa_mask = (const int*)d_in[1];
    const float* pair_act = (const float*)d_in[2];
    const float* qn_scale = (const float*)d_in[3];
    const float* qn_off   = (const float*)d_in[4];
    const float* pn_scale = (const float*)d_in[5];
    const float* pn_off   = (const float*)d_in[6];
    const float* w2d      = (const float*)d_in[7];
    const float* q_w      = (const float*)d_in[8];
    const float* k_w      = (const float*)d_in[9];
    const float* v_w      = (const float*)d_in[10];
    const float* o_w      = (const float*)d_in[11];
    const float* o_b      = (const float*)d_in[12];
    const float* gate_w   = (const float*)d_in[13];
    const float* gate_b   = (const float*)d_in[14];
    float* out = (float*)d_out;

    // workspace carve-up (bytes)
    char* ws = (char*)d_ws;
    const size_t SZ_ACT = 16777216ull * 2;        // 65536 x 256 bf16 = 32 MiB
    unsigned short* m_bf   = (unsigned short*)(ws);
    unsigned short* q_bf   = (unsigned short*)(ws + 1 * SZ_ACT);
    unsigned short* k_bf   = (unsigned short*)(ws + 2 * SZ_ACT);
    unsigned short* vt_bf  = (unsigned short*)(ws + 3 * SZ_ACT);
    unsigned short* g_bf   = (unsigned short*)(ws + 4 * SZ_ACT);
    unsigned short* wavg   = (unsigned short*)(ws + 5 * SZ_ACT);
    float*          nb     = (float*)(ws + 6 * SZ_ACT);              // 2 MiB
    unsigned short* qwT    = (unsigned short*)(ws + 6 * SZ_ACT + 2097152);
    unsigned short* kwT    = qwT + 65536;
    unsigned short* vwT    = kwT + 65536;
    unsigned short* gwT    = vwT + 65536;
    unsigned short* owT    = gwT + 65536;

    // K0: weight conversion (transposed bf16; q scaled by KEY_DIM^-0.5)
    wconv_kernel<<<1280, 256, 0, stream>>>(q_w, k_w, v_w, gate_w, o_w,
                                           qwT, kwT, vwT, gwT, owT);
    // K1/K2: LayerNorms (+ pair bias projection)
    ln_msa_kernel<<<8192, 256, 0, stream>>>(msa_act, qn_scale, qn_off, m_bf);
    ln_pair_kernel<<<8192, 256, 0, stream>>>(pair_act, pn_scale, pn_off, w2d, nb);

    // K3: Q/K/V/gate projections (WMMA bf16 GEMMs)
    dim3 ggrid(512, 16);
    proj_gemm_kernel<<<ggrid, 256, 0, stream>>>(m_bf, qwT, nullptr, q_bf, 0);
    proj_gemm_kernel<<<ggrid, 256, 0, stream>>>(m_bf, kwT, nullptr, k_bf, 1);
    proj_gemm_kernel<<<ggrid, 256, 0, stream>>>(m_bf, vwT, nullptr, vt_bf, 2);
    proj_gemm_kernel<<<ggrid, 256, 0, stream>>>(m_bf, gwT, gate_b, g_bf, 3);

    // K4: flash attention with pair bias + gating
    attn_kernel<<<4096, 256, 0, stream>>>(q_bf, k_bf, vt_bf, g_bf, nb,
                                          msa_mask, wavg);

    // K5: output projection + bias (fp32 result)
    out_gemm_kernel<<<ggrid, 256, 0, stream>>>(wavg, owT, o_b, out);
}
